// TemporalEvolutionLayer_54425825575261
// MI455X (gfx1250) — matrix operand order repaired and checked by reading, not compiled, not run on hardware
//
#include <hip/hip_runtime.h>
#include <hip/hip_bf16.h>

typedef __attribute__((ext_vector_type(16))) __bf16 v16bf;
typedef __attribute__((ext_vector_type(8)))  __bf16 v8bf;
typedef __attribute__((ext_vector_type(8)))  float  v8f;

#define TSTEPS 64
#define NBATCH 2048
#define DMOD   512
#define STR    520   // padded LDS row stride (elements) -> 1040B, 4-bank rotation

// Build a 16x32 bf16 A-fragment from LDS (row-major, padded rows).
// Per ISA layout: lane l holds row M=l&15; elements 0..7 = K kb+koff+0..7,
// elements 8..15 = K kb+16+koff+0..7 with koff = (l>>4)*8.
__device__ __forceinline__ v16bf afrag_lds(const __bf16* p) {
  v8bf lo = *(const v8bf*)p;
  v8bf hi = *(const v8bf*)(p + 16);
  return __builtin_shufflevector(lo, hi, 0,1,2,3,4,5,6,7,8,9,10,11,12,13,14,15);
}

__device__ __forceinline__ v8f wmma_bf16(v16bf a, v16bf b, v8f c) {
  return __builtin_amdgcn_wmma_f32_16x16x32_bf16(false, a, false, b, (short)0, c,
                                                 false, false);
}

// full-wave (32 lane) reduction of sum / sumsq
__device__ __forceinline__ void wred2(float& s, float& ss) {
#pragma unroll
  for (int o = 16; o >= 1; o >>= 1) {
    s += __shfl_xor(s, o, 32);
    ss += __shfl_xor(ss, o, 32);
  }
}
// half-wave (16 lane) reduction (xor masks < 16 keep halves separate)
__device__ __forceinline__ void hred2(float& s, float& ss) {
#pragma unroll
  for (int o = 8; o >= 1; o >>= 1) {
    s += __shfl_xor(s, o, 32);
    ss += __shfl_xor(ss, o, 32);
  }
}

// Pack fp32 weight [512 x cols] (row n = output, col k = input) into the
// bf16 B-fragment-native layout:
//   fragment (ntile nt, kstep ks): lane = (n&15) + (((k%32)/8)&1)*16,
//   element e = (k%8) + ((k%32)/16)*8;  lane data is 16 contiguous bf16 (32B).
__global__ void pack_weight(const float* __restrict__ W, __bf16* __restrict__ P,
                            int cols, int total) {
  int idx = blockIdx.x * blockDim.x + threadIdx.x;
  if (idx >= total) return;
  int n = idx / cols;
  int k = idx - n * cols;
  int nt = n >> 4, lN = n & 15;
  int ks = k >> 5, kr = k & 31;
  int lane = lN + (((kr >> 3) & 1) << 4);
  int e = (kr & 7) + ((kr >> 4) << 3);
  int ksteps = cols >> 5;
  size_t pidx = (((size_t)(nt * ksteps + ks) * 32) + lane) * 16 + e;
  P[pidx] = (__bf16)W[idx];
}

__global__ __launch_bounds__(1024)
void tel_main(const float* __restrict__ seq, const float* __restrict__ ts,
              const __bf16* __restrict__ WrP, const __bf16* __restrict__ WzP,
              const __bf16* __restrict__ WcP, const __bf16* __restrict__ WoP,
              const float* __restrict__ br, const float* __restrict__ bz,
              const float* __restrict__ bc, const float* __restrict__ bo,
              const float* __restrict__ gx, const float* __restrict__ bx,
              const float* __restrict__ gh, const float* __restrict__ bh,
              const float* __restrict__ gc, const float* __restrict__ bcell,
              const float* __restrict__ gf, const float* __restrict__ bfin,
              float* __restrict__ out) {
  // LDS: buf0 = xn / cell-LN'd state (time-multiplexed), hn, r*hn, reductions
  __shared__ __attribute__((aligned(16))) unsigned char smem[50176];
  __bf16* buf0 = (__bf16*)(smem);            // 16 x STR bf16  (16640 B)
  __bf16* hnS  = (__bf16*)(smem + 16640);    // 16 x STR bf16
  __bf16* rhS  = (__bf16*)(smem + 33280);    // 16 x STR bf16
  float*  redC = (float*)(smem + 49920);     // [0..15]=sum, [16..31]=sumsq (cell LN)
  float*  redO = redC + 32;                  // same, final LN

  const int tid  = threadIdx.x;
  const int wave = tid >> 5;
  const int lane = tid & 31;
  const int b0   = blockIdx.x * 16;

  const int lN    = lane & 15;
  const int hw    = lane >> 4;        // half-wave index
  const int nG    = wave * 16 + lN;   // this lane's output column (C layout: N=l&15)
  const int mBase = hw * 8;           // C layout: M = v + (lane>=16 ? 8 : 0)
  const int mA    = lN;               // A-frag row
  const int koffA = hw * 8;           // A-frag K sub-offset

  const float br_l = br[nG], bz_l = bz[nG], bc_l = bc[nG], bo_l = bo[nG];
  const float gc_l = gc[nG], bcl_l = bcell[nG];
  const float gf_l = gf[nG], bf_l = bfin[nG];

  for (int t = 0; t < TSTEPS; ++t) {
    // Opaque zero offset, refreshed every iteration: defeats LICM on the
    // t-invariant weight / LN-param loads (which previously spilled ~3.5KB/lane
    // to scratch) WITHOUT destroying pointer provenance, so the loads stay
    // global_load_b128 (global path, LOADcnt only) instead of flat_load_b128
    // (which would also tick DScnt and contend with the A-fragment ds traffic).
    size_t toff = 0;
    asm volatile("" : "+s"(toff));
    const __bf16 *WrQ = WrP + toff, *WzQ = WzP + toff;
    const __bf16 *WcQ = WcP + toff, *WoQ = WoP + toff;
    const float  *gxQ = gx + toff,  *bxQ = bx + toff;
    const float  *ghQ = gh + toff,  *bhQ = bh + toff;

    // ---- A1: waves 16..31 read previous state row from buf0, compute LN stats + decay
    float hv[16];
    float hmean = 0.f, hrstd = 0.f, decay = 0.f;
    if (wave >= 16 && t > 0) {
      const int m = wave - 16;
      float s = 0.f, ss = 0.f;
#pragma unroll
      for (int i = 0; i < 16; ++i) {
        float x = (float)buf0[m * STR + lane + 32 * i];
        hv[i] = x; s += x; ss += x * x;
      }
      wred2(s, ss);
      hmean = s * (1.f / 512.f);
      float var = ss * (1.f / 512.f) - hmean * hmean;
      hrstd = rsqrtf(var + 1e-5f);
      float dt = ts[(size_t)(b0 + m) * TSTEPS + t] -
                 ts[(size_t)(b0 + m) * TSTEPS + t - 1];
      dt = fminf(fmaxf(dt, 0.f), 10.f);
      decay = __expf(-dt);
    }
    if (tid < 64) redC[tid] = 0.f;  // zeroes redC[0..31] and redO[0..31]
    __syncthreads();

    // ---- A2: waves 0..15 write LN(x) into buf0; waves 16..31 write hn
    if (wave < 16) {
      const int m = wave;
      const float* xrow = seq + ((size_t)t * NBATCH + (b0 + m)) * DMOD;
      float xv[16];
      float s = 0.f, ss = 0.f;
#pragma unroll
      for (int i = 0; i < 16; ++i) {
        float x = xrow[lane + 32 * i];
        xv[i] = x; s += x; ss += x * x;
      }
      wred2(s, ss);
      float mean = s * (1.f / 512.f);
      float var  = ss * (1.f / 512.f) - mean * mean;
      float rstd = rsqrtf(var + 1e-5f);
#pragma unroll
      for (int i = 0; i < 16; ++i) {
        int e = lane + 32 * i;
        buf0[m * STR + e] = (__bf16)((xv[i] - mean) * rstd * gxQ[e] + bxQ[e]);
      }
    } else {
      const int m = wave - 16;
      if (t == 0) {
#pragma unroll
        for (int i = 0; i < 16; ++i) hnS[m * STR + lane + 32 * i] = (__bf16)0.f;
      } else {
#pragma unroll
        for (int i = 0; i < 16; ++i) {
          int e = lane + 32 * i;
          float hn = ((hv[i] - hmean) * hrstd * ghQ[e] + bhQ[e]) * decay;
          hnS[m * STR + e] = (__bf16)hn;
        }
      }
    }
    __syncthreads();

    // ---- B: r and z gate GEMMs over K = [xn | hn] (1024)
    v8f racc = {}, zacc = {};
    for (int ks = 0; ks < 32; ++ks) {
      const int kb = ks * 32;
      const __bf16* ap = (kb < 512) ? (buf0 + mA * STR + kb + koffA)
                                    : (hnS + mA * STR + (kb - 512) + koffA);
      v16bf a = afrag_lds(ap);
      v16bf bR = *(const v16bf*)(WrQ + (((size_t)wave * 32 + ks) * 32 + lane) * 16);
      v16bf bZ = *(const v16bf*)(WzQ + (((size_t)wave * 32 + ks) * 32 + lane) * 16);
      racc = wmma_bf16(a, bR, racc);
      zacc = wmma_bf16(a, bZ, zacc);
    }
    float rv[8], zv[8];
#pragma unroll
    for (int v = 0; v < 8; ++v) {
      rv[v] = 1.f / (1.f + __expf(-(racc[v] + br_l)));
      zv[v] = 1.f / (1.f + __expf(-(zacc[v] + bz_l)));
    }
#pragma unroll
    for (int v = 0; v < 8; ++v) {
      int m = mBase + v;
      float hn = (float)hnS[m * STR + nG];
      rhS[m * STR + nG] = (__bf16)(rv[v] * hn);
    }
    __syncthreads();

    // ---- C: candidate GEMM over K = [xn | r*hn], then h_new + cell-LN partials
    v8f cacc = {};
    for (int ks = 0; ks < 32; ++ks) {
      const int kb = ks * 32;
      const __bf16* ap = (kb < 512) ? (buf0 + mA * STR + kb + koffA)
                                    : (rhS + mA * STR + (kb - 512) + koffA);
      v16bf a = afrag_lds(ap);
      v16bf bC = *(const v16bf*)(WcQ + (((size_t)wave * 32 + ks) * 32 + lane) * 16);
      cacc = wmma_bf16(a, bC, cacc);
    }
    float hnew[8];
#pragma unroll
    for (int v = 0; v < 8; ++v) {
      int m = mBase + v;
      float ht = tanhf(cacc[v] + bc_l);
      float hn = (float)hnS[m * STR + nG];
      hnew[v] = (1.f - zv[v]) * hn + zv[v] * ht;
      float s = hnew[v], ss = hnew[v] * hnew[v];
      hred2(s, ss);
      if (lN == 0) {
        atomicAdd(&redC[m], s);
        atomicAdd(&redC[16 + m], ss);
      }
    }
    __syncthreads();

    // ---- D: cell layer-norm -> new state (bf16 into buf0; xn is dead now)
#pragma unroll
    for (int v = 0; v < 8; ++v) {
      int m = mBase + v;
      float mean = redC[m] * (1.f / 512.f);
      float var  = redC[16 + m] * (1.f / 512.f) - mean * mean;
      float rstd = rsqrtf(var + 1e-5f);
      buf0[m * STR + nG] = (__bf16)((hnew[v] - mean) * rstd * gc_l + bcl_l);
    }
    __syncthreads();

    // ---- E: output projection (K=512) + bias + residual, final-LN partials
    v8f oacc = {};
    for (int ks = 0; ks < 16; ++ks) {
      const int kb = ks * 32;
      v16bf a = afrag_lds(buf0 + mA * STR + kb + koffA);
      v16bf bO = *(const v16bf*)(WoQ + (((size_t)wave * 16 + ks) * 32 + lane) * 16);
      oacc = wmma_bf16(a, bO, oacc);
    }
    float opre[8];
#pragma unroll
    for (int v = 0; v < 8; ++v) {
      int m = mBase + v;
      float x = oacc[v] + bo_l + seq[((size_t)t * NBATCH + (b0 + m)) * DMOD + nG];
      opre[v] = x;
      float s = x, ss = x * x;
      hred2(s, ss);
      if (lN == 0) {
        atomicAdd(&redO[m], s);
        atomicAdd(&redO[16 + m], ss);
      }
    }
    __syncthreads();

    // ---- F: final layer-norm + store
#pragma unroll
    for (int v = 0; v < 8; ++v) {
      int m = mBase + v;
      float mean = redO[m] * (1.f / 512.f);
      float var  = redO[16 + m] * (1.f / 512.f) - mean * mean;
      float rstd = rsqrtf(var + 1e-5f);
      out[((size_t)t * NBATCH + (b0 + m)) * DMOD + nG] =
          (opre[v] - mean) * rstd * gf_l + bf_l;
    }
    __syncthreads();
  }
}

extern "C" void kernel_launch(void* const* d_in, const int* in_sizes, int n_in,
                              void* d_out, int out_size, void* d_ws, size_t ws_size,
                              hipStream_t stream) {
  const float* seq   = (const float*)d_in[0];
  const float* ts    = (const float*)d_in[1];
  const float* Wr    = (const float*)d_in[2];
  const float* br    = (const float*)d_in[3];
  const float* Wz    = (const float*)d_in[4];
  const float* bz    = (const float*)d_in[5];
  const float* Wc    = (const float*)d_in[6];
  const float* bc    = (const float*)d_in[7];
  const float* Wo    = (const float*)d_in[8];
  const float* bo    = (const float*)d_in[9];
  const float* gx    = (const float*)d_in[10];
  const float* bx    = (const float*)d_in[11];
  const float* gh    = (const float*)d_in[12];
  const float* bh    = (const float*)d_in[13];
  const float* gc    = (const float*)d_in[14];
  const float* bcell = (const float*)d_in[15];
  const float* gf    = (const float*)d_in[16];
  const float* bfin  = (const float*)d_in[17];
  float* out = (float*)d_out;

  // bf16 packed weights in workspace: Wr, Wz, Wc (512x1024) + Wo (512x512) = 3.5 MB
  __bf16* WrP = (__bf16*)d_ws;
  __bf16* WzP = WrP + 512 * 1024;
  __bf16* WcP = WzP + 512 * 1024;
  __bf16* WoP = WcP + 512 * 1024;

  const int tgate = 512 * 1024;
  const int tout  = 512 * 512;
  pack_weight<<<(tgate + 255) / 256, 256, 0, stream>>>(Wr, WrP, 1024, tgate);
  pack_weight<<<(tgate + 255) / 256, 256, 0, stream>>>(Wz, WzP, 1024, tgate);
  pack_weight<<<(tgate + 255) / 256, 256, 0, stream>>>(Wc, WcP, 1024, tgate);
  pack_weight<<<(tout + 255) / 256, 256, 0, stream>>>(Wo, WoP, 512, tout);

  tel_main<<<NBATCH / 16, 1024, 0, stream>>>(seq, ts, WrP, WzP, WcP, WoP,
                                             br, bz, bc, bo, gx, bx, gh, bh,
                                             gc, bcell, gf, bfin, out);
}
